// KSparseAutoencoder_41291815584089
// MI455X (gfx1250) — compile-verified, matrix operand order checked by reading, not run on hardware
//
#include <hip/hip_runtime.h>
#include <hip/hip_bf16.h>

// ---------------------------------------------------------------------------
// K-sparse autoencoder for MI455X (gfx1250, wave32, WMMA bf16 + TDM staging)
//   z     = relu(x @ W_enc + b_enc)        [8192, 16384]
//   mask  = top-64 per row of z
//   x_hat = (z*mask) @ W_dec + b_dec       [8192, 1536]
//   d_out = [x_hat | z*mask]
// ---------------------------------------------------------------------------

#define BATCH 8192
#define IDIM  1536
#define HDIM  16384
#define TOPK  64

typedef __attribute__((ext_vector_type(16))) __bf16 v16bf;
typedef __attribute__((ext_vector_type(8)))  float  v8f;
typedef __attribute__((ext_vector_type(4)))  unsigned int u32x4;
typedef __attribute__((ext_vector_type(8)))  int i32x8;
typedef __attribute__((ext_vector_type(4)))  int i32x4;

union ABFrag {
    u32x4 q[2];   // two ds_load_b128 worth of bf16 data
    v16bf v;      // WMMA operand
};

__device__ __forceinline__ unsigned short f2bf(float f) {
    unsigned u = __float_as_uint(f);
    unsigned r = u + 0x7FFFu + ((u >> 16) & 1u);   // round-to-nearest-even
    return (unsigned short)(r >> 16);
}

// ---------------------------------------------------------------------------
// Kernel 1: x (f32) -> bf16, 8 elements / thread
// ---------------------------------------------------------------------------
__global__ __launch_bounds__(256)
void k_convert_x(const float* __restrict__ x, unsigned short* __restrict__ xbf) {
    const size_t i = ((size_t)blockIdx.x * 256 + threadIdx.x) * 8;
    float4 f0 = *(const float4*)(x + i);
    float4 f1 = *(const float4*)(x + i + 4);
    uint4 o;
    o.x = (unsigned)f2bf(f0.x) | ((unsigned)f2bf(f0.y) << 16);
    o.y = (unsigned)f2bf(f0.z) | ((unsigned)f2bf(f0.w) << 16);
    o.z = (unsigned)f2bf(f1.x) | ((unsigned)f2bf(f1.y) << 16);
    o.w = (unsigned)f2bf(f1.z) | ((unsigned)f2bf(f1.w) << 16);
    *(uint4*)(xbf + i) = o;
}

// ---------------------------------------------------------------------------
// Kernel 2: W_enc [IDIM][HDIM] f32 -> W_T [HDIM][IDIM] bf16 (LDS tiled 32x32)
// ---------------------------------------------------------------------------
__global__ __launch_bounds__(256)
void k_transpose_w(const float* __restrict__ W, unsigned short* __restrict__ WT) {
    __shared__ unsigned short tile[32][33];
    const int tx = threadIdx.x & 31;
    const int ty = threadIdx.x >> 5;               // 0..7
    const int n0 = blockIdx.x * 32;                // hidden dim
    const int k0 = blockIdx.y * 32;                // input dim
#pragma unroll
    for (int i = 0; i < 4; ++i) {
        const int k = k0 + ty + i * 8;
        tile[ty + i * 8][tx] = f2bf(W[(size_t)k * HDIM + n0 + tx]);
    }
    __syncthreads();
#pragma unroll
    for (int i = 0; i < 4; ++i) {
        const int n = n0 + ty + i * 8;
        WT[(size_t)n * IDIM + k0 + tx] = tile[tx][ty + i * 8];
    }
}

// ---------------------------------------------------------------------------
// Kernel 3: encoder GEMM via v_wmma_f32_16x16x32_bf16
//   block tile 128(M) x 128(N), 8 waves, each wave 32x64 (2x4 WMMA tiles)
//   K-loop step 32, double-buffered LDS, rows padded to 40 shorts (80 B)
//   so the per-fragment ds_load_b128 pattern is bank-conflict free.
//   Staging: Tensor Data Mover (tensor_load_to_lds); the TDM descriptor's pad
//   fields (16 DWORDs data + 4 DWORDs pad) reproduce the 80-byte LDS stride.
//   The TDM builtin writes LDS invisibly to the compiler, so fragment loads
//   are explicit asm ds_load_b128 + an s_wait_dscnt that ties the registers.
// ---------------------------------------------------------------------------
#define LDSW 40   // padded row stride in bf16 units (80 bytes)
#define TILE_BYTES (128 * LDSW * 2)   // 10240 B per tile buffer

#if defined(__gfx1250__) && __has_builtin(__builtin_amdgcn_tensor_load_to_lds)
#define USE_TDM 1
#else
#define USE_TDM 0
#endif

struct SmemT {
    unsigned short A[2][128 * LDSW];
    unsigned short B[2][128 * LDSW];
};

#if USE_TDM
// DMA one 128-row x 32-bf16 tile (global row stride IDIM elements) into LDS
// at byte offset lds_off, padding each 64 B row to 80 B.
__device__ __forceinline__ void tdm_load_tile(unsigned lds_off,
                                              unsigned long long gaddr) {
    u32x4 g0;
    g0.x = 1u;                                         // count=1, user mode
    g0.y = lds_off;                                    // lds_addr (bytes)
    g0.z = (unsigned)(gaddr & 0xFFFFFFFFull);          // global_addr[31:0]
    g0.w = (unsigned)((gaddr >> 32) & 0x1FFFFFFull)    // global_addr[56:32]
         | (2u << 30);                                 // type=2 (image)
    i32x8 g1;
    g1[0] = (int)((1u << 16)      // data_size = 2 bytes (bf16)
                | (1u << 20)      // pad_enable
                | (3u << 22)      // pad_interval: 16 DWORDs (=64 B row)
                | (3u << 25));    // pad_amount:   4 DWORDs (=16 B pad)
    g1[1] = (int)(((unsigned)IDIM & 0xFFFFu) << 16);   // tensor_dim0 lo16
    g1[2] = (int)((((unsigned)IDIM >> 16) & 0xFFFFu)   // tensor_dim0 hi16
                | (((unsigned)HDIM & 0xFFFFu) << 16)); // tensor_dim1 lo16
    g1[3] = (int)((((unsigned)HDIM >> 16) & 0xFFFFu)   // tensor_dim1 hi16
                | (32u << 16));                        // tile_dim0 = 32
    g1[4] = (int)(128u);                               // tile_dim1 = 128, tile_dim2 = 0
    g1[5] = (int)(unsigned)IDIM;                       // tensor_dim0_stride lo32
    g1[6] = 0;                                         // stride hi16 | dim1_stride lo16
    g1[7] = 0;
    const i32x4 z4 = {0, 0, 0, 0};
#if __clang_major__ >= 23
    const i32x8 z8 = {0, 0, 0, 0, 0, 0, 0, 0};
    __builtin_amdgcn_tensor_load_to_lds(g0, g1, z4, z4, z8, 0);
#else
    __builtin_amdgcn_tensor_load_to_lds(g0, g1, z4, z4, 0);
#endif
}

// Two ds_load_b128 of one 16-bf16 WMMA fragment (chunks at +0 and +32 B).
__device__ __forceinline__ void lds_load_frag(ABFrag& f, unsigned byte_addr) {
    asm volatile("ds_load_b128 %0, %2\n\t"
                 "ds_load_b128 %1, %2 offset:32"
                 : "=v"(f.q[0]), "=v"(f.q[1])
                 : "v"(byte_addr));
}
#endif

__global__ __launch_bounds__(256)
void k_encoder_wmma(const unsigned short* __restrict__ xbf,
                    const unsigned short* __restrict__ wT,
                    const float* __restrict__ benc,
                    float* __restrict__ z) {
    __shared__ __align__(16) SmemT smem;

    const int tid   = threadIdx.x;
    const int lane  = tid & 31;
    const int wave  = tid >> 5;
    const int half  = lane >> 4;       // 0|1 -> selects K sub-range per WMMA layout
    const int l15   = lane & 15;
    const int waveM = wave & 3;        // 4 waves along M  -> 4*32 = 128
    const int waveN = wave >> 2;       // 2 waves along N  -> 2*64 = 128
    const int bm = blockIdx.y * 128;
    const int bn = blockIdx.x * 128;

    v8f acc[2][4] = {};
    const int KTILES = IDIM / 32;   // 48

#if USE_TDM
    // Escape smem's address so the LDS object stays allocated even though all
    // writes (TDM) and reads (asm ds_load) are invisible to the optimizer.
    {
        void* pKeep = (void*)&smem;
        asm volatile("" : "+v"(pKeep) :: "memory");
    }
    // smem is the only LDS object -> struct sits at LDS offset 0.
    const unsigned long long gA = (unsigned long long)(uintptr_t)(xbf + (size_t)bm * IDIM);
    const unsigned long long gB = (unsigned long long)(uintptr_t)(wT  + (size_t)bn * IDIM);
    if (wave == 0) {
        tdm_load_tile(0u,              gA);            // A tile 0 -> buf 0
        tdm_load_tile(2u * TILE_BYTES, gB);            // B tile 0 -> buf 0
    }
    for (int kt = 0; kt < KTILES; ++kt) {
        const int cur = kt & 1;
        if (wave == 0) {
            if (kt + 1 < KTILES) {
                const unsigned long long koff = (unsigned long long)(kt + 1) * 64u; // 32 bf16
                tdm_load_tile((unsigned)(cur ^ 1) * TILE_BYTES,                   gA + koff);
                tdm_load_tile(2u * TILE_BYTES + (unsigned)(cur ^ 1) * TILE_BYTES, gB + koff);
                __builtin_amdgcn_s_wait_tensorcnt(2);  // in-order: tile `cur` done
            } else {
                __builtin_amdgcn_s_wait_tensorcnt(0);
            }
        }
        __syncthreads();   // tile `cur` visible to all waves

        ABFrag a[2], b[4];
        const unsigned baseA = (unsigned)cur * TILE_BYTES;
        const unsigned baseB = 2u * TILE_BYTES + (unsigned)cur * TILE_BYTES;
#pragma unroll
        for (int mt = 0; mt < 2; ++mt) {
            const int r = waveM * 32 + mt * 16 + l15;
            lds_load_frag(a[mt], baseA + (unsigned)(r * LDSW + half * 8) * 2u);
        }
#pragma unroll
        for (int nt = 0; nt < 4; ++nt) {
            const int r = waveN * 64 + nt * 16 + l15;
            lds_load_frag(b[nt], baseB + (unsigned)(r * LDSW + half * 8) * 2u);
        }
        // One wait for all 12 ds_loads; tying the registers forces every
        // consumer (the WMMAs) to be scheduled after the wait.
        asm volatile("s_wait_dscnt 0"
            : "+v"(a[0].q[0]), "+v"(a[0].q[1]), "+v"(a[1].q[0]), "+v"(a[1].q[1]),
              "+v"(b[0].q[0]), "+v"(b[0].q[1]), "+v"(b[1].q[0]), "+v"(b[1].q[1]),
              "+v"(b[2].q[0]), "+v"(b[2].q[1]), "+v"(b[3].q[0]), "+v"(b[3].q[1])
            :: "memory");

#pragma unroll
        for (int mt = 0; mt < 2; ++mt)
#pragma unroll
            for (int nt = 0; nt < 4; ++nt)
                acc[mt][nt] = __builtin_amdgcn_wmma_f32_16x16x32_bf16(
                    false, a[mt].v, false, b[nt].v,
                    (short)0, acc[mt][nt], false, false);

        __syncthreads();   // DScnt==0 here; everyone done with `cur` before TDM reuses it
    }
#else
    // Fallback: VGPR staging (also what the host-side compile pass sees)
    auto stage = [&](int buf, int kt) {
        const int k0 = kt * 32;
#pragma unroll
        for (int i = 0; i < 2; ++i) {
            const int c   = tid + i * 256;   // 512 16-byte chunks per operand
            const int row = c >> 2;          // 0..127
            const int ko  = (c & 3) * 8;     // 0,8,16,24 bf16
            *(u32x4*)&smem.A[buf][row * LDSW + ko] =
                *(const u32x4*)(xbf + (size_t)(bm + row) * IDIM + k0 + ko);
            *(u32x4*)&smem.B[buf][row * LDSW + ko] =
                *(const u32x4*)(wT + (size_t)(bn + row) * IDIM + k0 + ko);
        }
    };

    stage(0, 0);
    __syncthreads();
    for (int kt = 0; kt < KTILES; ++kt) {
        const int cur = kt & 1;
        if (kt + 1 < KTILES) stage(cur ^ 1, kt + 1);

        ABFrag a[2], b[4];
#pragma unroll
        for (int mt = 0; mt < 2; ++mt) {
            const int r    = waveM * 32 + mt * 16 + l15;
            const int base = r * LDSW + half * 8;
            a[mt].q[0] = *(const u32x4*)&smem.A[cur][base];
            a[mt].q[1] = *(const u32x4*)&smem.A[cur][base + 16];
        }
#pragma unroll
        for (int nt = 0; nt < 4; ++nt) {
            const int r    = waveN * 64 + nt * 16 + l15;
            const int base = r * LDSW + half * 8;
            b[nt].q[0] = *(const u32x4*)&smem.B[cur][base];
            b[nt].q[1] = *(const u32x4*)&smem.B[cur][base + 16];
        }
#pragma unroll
        for (int mt = 0; mt < 2; ++mt)
#pragma unroll
            for (int nt = 0; nt < 4; ++nt)
                acc[mt][nt] = __builtin_amdgcn_wmma_f32_16x16x32_bf16(
                    false, a[mt].v, false, b[nt].v,
                    (short)0, acc[mt][nt], false, false);

        __syncthreads();
    }
#endif

    // epilogue: bias + relu, f32 store
#pragma unroll
    for (int nt = 0; nt < 4; ++nt) {
        const int col = bn + waveN * 64 + nt * 16 + l15;
        const float bv = benc[col];
#pragma unroll
        for (int mt = 0; mt < 2; ++mt) {
#pragma unroll
            for (int r = 0; r < 8; ++r) {
                const int row = bm + waveM * 32 + mt * 16 + r + half * 8;
                float v = acc[mt][nt][r] + bv;
                z[(size_t)row * HDIM + col] = v > 0.0f ? v : 0.0f;
            }
        }
    }
}

// ---------------------------------------------------------------------------
// Kernel 4: per-row top-64 radix select (values are relu'd => >= 0, so uint
// bit order == float order). One 256-thread block per row. High 16 bits live
// in 32 KB LDS for passes 0-1; passes 2-3 and the rewrite re-read the row
// (block's own 64 KB -> L2 hit). Writes masked z in place + compact lists.
// ---------------------------------------------------------------------------
__global__ __launch_bounds__(256)
void k_topk(float* __restrict__ z,
            float* __restrict__ tv, int* __restrict__ ti) {
    __shared__ unsigned short hi16[HDIM];
    __shared__ int hist[256];
    __shared__ unsigned sPrefix;
    __shared__ int sRem, sTie, sSlot;

    const int row = blockIdx.x;
    const int tid = threadIdx.x;
    float* zr = z + (size_t)row * HDIM;

    for (int j = tid; j < HDIM; j += 256)
        hi16[j] = (unsigned short)(__float_as_uint(zr[j]) >> 16);
    if (tid == 0) { sPrefix = 0u; sRem = TOPK; }

    for (int p = 0; p < 4; ++p) {
        hist[tid] = 0;
        __syncthreads();
        const unsigned pref  = sPrefix;
        const int      shift = 24 - 8 * p;
        for (int j = tid; j < HDIM; j += 256) {
            unsigned u = (p < 2) ? ((unsigned)hi16[j] << 16) : __float_as_uint(zr[j]);
            if (p == 0 || (u >> (shift + 8)) == pref)
                atomicAdd(&hist[(u >> shift) & 0xFF], 1);
        }
        __syncthreads();
        if (tid == 0) {
            int rem = sRem, cum = 0, sel = 0;
            for (int b = 255; b >= 0; --b) {
                const int c = hist[b];
                if (cum + c >= rem) { sel = b; rem -= cum; break; }
                cum += c;
            }
            sPrefix = (pref << 8) | (unsigned)sel;
            sRem = rem;
        }
        __syncthreads();
    }

    if (tid == 0) { sTie = 0; sSlot = 0; }
    __syncthreads();
    const unsigned T   = sPrefix;   // bits of the K-th largest value
    const int      rem = sRem;      // how many ties at T to keep

    for (int j = tid; j < HDIM; j += 256) {
        const unsigned u = __float_as_uint(zr[j]);
        bool keep = (u > T);
        if (!keep && u == T) keep = (atomicAdd(&sTie, 1) < rem);
        const float v = keep ? __uint_as_float(u) : 0.0f;
        zr[j] = v;
        if (keep) {
            const int s = atomicAdd(&sSlot, 1);
            if (s < TOPK) {
                tv[(size_t)row * TOPK + s] = v;
                ti[(size_t)row * TOPK + s] = j;
            }
        }
    }
}

// ---------------------------------------------------------------------------
// Kernel 5: sparse decode  x_hat[b,:] = b_dec + sum_j tv[b,j] * W_dec[ti[b,j],:]
// W_dec (100 MB) is L2-resident; one block per batch row, 6 cols/thread.
// ---------------------------------------------------------------------------
__global__ __launch_bounds__(256)
void k_decode(const float* __restrict__ tv, const int* __restrict__ ti,
              const float* __restrict__ Wdec, const float* __restrict__ bdec,
              float* __restrict__ xhat) {
    __shared__ float sv[TOPK];
    __shared__ int   si[TOPK];
    const int row = blockIdx.x;
    const int tid = threadIdx.x;
    if (tid < TOPK) {
        sv[tid] = tv[(size_t)row * TOPK + tid];
        si[tid] = ti[(size_t)row * TOPK + tid];
    }
    __syncthreads();

    float acc[6];
#pragma unroll
    for (int i = 0; i < 6; ++i) acc[i] = bdec[tid + i * 256];

    for (int j = 0; j < TOPK; ++j) {
        const float v = sv[j];
        const float* wr = Wdec + (size_t)si[j] * IDIM;
        if (j + 1 < TOPK)
            __builtin_prefetch(Wdec + (size_t)si[j + 1] * IDIM + tid, 0, 1);
#pragma unroll
        for (int i = 0; i < 6; ++i) acc[i] += v * wr[tid + i * 256];
    }
#pragma unroll
    for (int i = 0; i < 6; ++i)
        xhat[(size_t)row * IDIM + tid + i * 256] = acc[i];
}

// ---------------------------------------------------------------------------
extern "C" void kernel_launch(void* const* d_in, const int* in_sizes, int n_in,
                              void* d_out, int out_size, void* d_ws, size_t ws_size,
                              hipStream_t stream) {
    const float* x    = (const float*)d_in[0];
    const float* Wenc = (const float*)d_in[1];
    const float* benc = (const float*)d_in[2];
    const float* Wdec = (const float*)d_in[3];
    const float* bdec = (const float*)d_in[4];

    float* xhat = (float*)d_out;                         // [BATCH, IDIM]
    float* z    = (float*)d_out + (size_t)BATCH * IDIM;  // [BATCH, HDIM]

    // workspace layout
    unsigned short* xbf = (unsigned short*)d_ws;                 // BATCH*IDIM bf16
    unsigned short* wT  = xbf + (size_t)BATCH * IDIM;            // HDIM*IDIM bf16
    float* tv = (float*)(wT + (size_t)HDIM * IDIM);              // BATCH*TOPK f32
    int*   ti = (int*)(tv + (size_t)BATCH * TOPK);               // BATCH*TOPK i32

    k_convert_x<<<(BATCH * IDIM) / (256 * 8), 256, 0, stream>>>(x, xbf);
    k_transpose_w<<<dim3(HDIM / 32, IDIM / 32), 256, 0, stream>>>(Wenc, wT);
    k_encoder_wmma<<<dim3(HDIM / 128, BATCH / 128), 256, 0, stream>>>(xbf, wT, benc, z);
    k_topk<<<BATCH, 256, 0, stream>>>(z, tv, ti);
    k_decode<<<BATCH, 256, 0, stream>>>(tv, ti, Wdec, bdec, xhat);
}
